// ntn_layer_70394513981736
// MI455X (gfx1250) — matrix operand order repaired and checked by reading, not compiled, not run on hardware
//
#include <hip/hip_runtime.h>
#include <math.h>

typedef __attribute__((ext_vector_type(16))) __bf16 v16bf;
typedef __attribute__((ext_vector_type(8)))  float  v8f;
typedef __attribute__((ext_vector_type(4)))  float  float4v;

#define DIM 256
#define STAGE_BYTES 8192

#define WMMA_BF16(A, Bf, C) \
  __builtin_amdgcn_wmma_f32_16x16x32_bf16(false, (A), false, (Bf), (short)0, (C), false, false)

static __device__ __forceinline__ __bf16 f2bf(float f) { return (__bf16)f; }

#define AS3 __attribute__((address_space(3)))

#if __has_builtin(__builtin_amdgcn_sched_barrier)
#define SCHED_FENCE() __builtin_amdgcn_sched_barrier(0)
#else
#define SCHED_FENCE() asm volatile("" ::: "memory")
#endif

// LDS byte offset of a generic pointer into shared memory.
static __device__ __forceinline__ unsigned lds_off(const void* p) {
  return (unsigned)(size_t)(AS3 const void*)(p);
}

// CDNA5 async global->LDS copy, 16 bytes per lane, tracked by ASYNCcnt.
static __device__ __forceinline__ void async_b128(const void* gsrc, unsigned dst_lds_off) {
  asm volatile("global_load_async_to_lds_b128 %0, %1, off"
               :: "v"(dst_lds_off), "v"(gsrc) : "memory");
}
static __device__ __forceinline__ void wait_async0() {
  asm volatile("s_wait_asynccnt 0x0" ::: "memory");
}

// ---------------------------------------------------------------------------
// Pack W [16,256,256] f32 -> bf16 fragments in WMMA B-matrix layout.
// Fragment order: (k, ntile, kk); within fragment lane-major, 16 bf16/lane.
// B-layout (32x16, 16-bit): col = lane&15, K = kk*32 + (lane>=16?8:0) + i + (i>=8?8:0).
// ---------------------------------------------------------------------------
__global__ void pack_w_kernel(const float* __restrict__ W, __bf16* __restrict__ dst) {
  unsigned tid = blockIdx.x * blockDim.x + threadIdx.x;   // 0 .. 1048575
  unsigned i    =  tid        & 15u;
  unsigned lane = (tid >> 4)  & 31u;
  unsigned kk   = (tid >> 9)  & 7u;
  unsigned nt   = (tid >> 12) & 15u;
  unsigned k    =  tid >> 16;
  unsigned d = kk * 32u + ((lane >= 16u) ? 8u : 0u) + i + ((i >= 8u) ? 8u : 0u);
  unsigned e = nt * 16u + (lane & 15u);
  dst[tid] = f2bf(W[((size_t)k * 256u + d) * 256u + e]);
}

// Pack V and G [512,16] f32 -> bf16 B-fragments (16 K-chunks, single N-tile).
__global__ void pack_vg_kernel(const float* __restrict__ V, const float* __restrict__ G,
                               __bf16* __restrict__ dv, __bf16* __restrict__ dg) {
  unsigned tid = blockIdx.x * blockDim.x + threadIdx.x;   // 0 .. 8191
  unsigned i    =  tid       & 15u;
  unsigned lane = (tid >> 4) & 31u;
  unsigned kk   =  tid >> 9;                              // 0..15
  unsigned row = kk * 32u + ((lane >= 16u) ? 8u : 0u) + i + ((i >= 8u) ? 8u : 0u);
  unsigned col = lane & 15u;
  dv[tid] = f2bf(V[row * 16u + col]);
  dg[tid] = f2bf(G[row * 16u + col]);
}

// ---------------------------------------------------------------------------
// Fused NTN kernel: 8 waves/block, one 16-row batch tile per wave.
// Packed W streamed once per block through double-buffered LDS staging
// (async-to-LDS), shared by all 8 waves -> 8x less L2 traffic.
// ---------------------------------------------------------------------------
__global__ __launch_bounds__(256) void ntn_main_kernel(
    const float* __restrict__ e1, const float* __restrict__ e2,
    const __bf16* __restrict__ pw,
    const __bf16* __restrict__ pv,
    const __bf16* __restrict__ pg,
    const float* __restrict__ b1, const float* __restrict__ b2,
    float* __restrict__ out) {
  extern __shared__ char smem[];
  char*  stage = smem;                                   // 2 x 8 KB W staging
  float* e2c   = (float*)(smem + 2 * STAGE_BYTES);       // 8 waves x 16 KB e2 tile

  const int tid  = (int)threadIdx.x;
  const int wave = tid >> 5;
  const int lane = tid & 31;
  const int g    = lane >> 4;       // half-wave group
  const int n    = lane & 15;       // N/column index (k-index for V/G accs)
  const int m0   = (int)blockIdx.x * 128 + wave * 16;

  // Kick off staging of chunk 0 immediately; it lands while phase 1 runs.
  const unsigned stage_off = lds_off(stage);
  async_b128((const char*)pw + tid * 16,        stage_off + tid * 16);
  async_b128((const char*)pw + 4096 + tid * 16, stage_off + 4096 + tid * 16);

  const float* e1row = e1 + (size_t)(m0 + n) * DIM;      // A-layout: M = lane&15
  const float* e2row = e2 + (size_t)(m0 + n) * DIM;

  // -------- Phase 1: A fragments + fused forward/match GEMMs (32 WMMAs) ---
  v8f accF = {};
  v8f accM = {};
  v16bf a[8];
#pragma unroll
  for (int kk = 0; kk < 8; ++kk) {
    const int off1 = kk * 32 + 8 * g;   // K runs: [off1,off1+8), [off1+16,off1+24)
    float x1[16], x2[16];
    *(float4v*)&x1[0]  = *(const float4v*)(e1row + off1);
    *(float4v*)&x1[4]  = *(const float4v*)(e1row + off1 + 4);
    *(float4v*)&x1[8]  = *(const float4v*)(e1row + off1 + 16);
    *(float4v*)&x1[12] = *(const float4v*)(e1row + off1 + 20);
    *(float4v*)&x2[0]  = *(const float4v*)(e2row + off1);
    *(float4v*)&x2[4]  = *(const float4v*)(e2row + off1 + 4);
    *(float4v*)&x2[8]  = *(const float4v*)(e2row + off1 + 16);
    *(float4v*)&x2[12] = *(const float4v*)(e2row + off1 + 20);

    v16bf u1, u2, ud, up;
#pragma unroll
    for (int i = 0; i < 16; ++i) {
      const float f1 = x1[i], f2 = x2[i];
      u1[i] = f2bf(f1);
      u2[i] = f2bf(f2);
      ud[i] = f2bf(fabsf(f2 - f1));
      up[i] = f2bf(f1 * f2);
    }
    a[kk] = u1;

    const v16bf vf0 = *(const v16bf*)(pv + (size_t)kk * 512 + lane * 16);
    const v16bf vf1 = *(const v16bf*)(pv + (size_t)(kk + 8) * 512 + lane * 16);
    const v16bf gf0 = *(const v16bf*)(pg + (size_t)kk * 512 + lane * 16);
    const v16bf gf1 = *(const v16bf*)(pg + (size_t)(kk + 8) * 512 + lane * 16);
    accF = WMMA_BF16(u1, vf0, accF);   // concat([e1,e2]) @ V : e1 half
    accF = WMMA_BF16(u2, vf1, accF);   //                      e2 half
    accM = WMMA_BF16(ud, gf0, accM);   // concat([|e2-e1|, e1*e2]) @ G
    accM = WMMA_BF16(up, gf1, accM);
    SCHED_FENCE();   // cap liveness: keep one iteration's temps in flight
  }

  // -------- e2 tile in C-matrix layout, f32 in this wave's LDS slice ------
  // Read layout per (nt): 8 contiguous floats at (nt*32+lane)*8.
  float* e2w = e2c + wave * 4096;
#pragma unroll
  for (int nt = 0; nt < 16; ++nt) {
    float4v w0, w1;
#pragma unroll
    for (int r = 0; r < 4; ++r) {
      w0[r] = e2[(size_t)(m0 + 8 * g + r)     * DIM + nt * 16 + n];
      w1[r] = e2[(size_t)(m0 + 8 * g + 4 + r) * DIM + nt * 16 + n];
    }
    *(float4v*)(e2w + ((size_t)nt * 32 + lane) * 8)     = w0;
    *(float4v*)(e2w + ((size_t)nt * 32 + lane) * 8 + 4) = w1;
    SCHED_FENCE();
  }

  // -------- Phase 2: bilinear term over 256 staged (k,ntile) chunks -------
  float tsel[8];
#pragma unroll
  for (int r = 0; r < 8; ++r) tsel[r] = 0.f;

#pragma unroll 1
  for (int k = 0; k < 16; ++k) {
    float acc_t[8];
#pragma unroll
    for (int r = 0; r < 8; ++r) acc_t[r] = 0.f;

#pragma unroll 1
    for (int nt = 0; nt < 16; ++nt) {
      const int c = k * 16 + nt;

      wait_async0();        // this wave's staged loads landed
      __syncthreads();      // all waves' loads landed; prior compute done

      // Prefetch next chunk into the other buffer. Index clamps to chunk 0 on
      // the final iteration: the barrier above guarantees every wave finished
      // reading that buffer (chunk c-1), so the redundant re-stage is benign.
      const int cn = (c + 1) & 255;
      const char* gsrc = (const char*)pw + (size_t)cn * STAGE_BYTES;
      const unsigned dsto = stage_off + (unsigned)(cn & 1) * STAGE_BYTES;
      async_b128(gsrc + tid * 16,        dsto + tid * 16);
      async_b128(gsrc + 4096 + tid * 16, dsto + 4096 + tid * 16);

      const char* buf = stage + (unsigned)(c & 1) * STAGE_BYTES;
      v8f acc = {};
#pragma unroll
      for (int kk = 0; kk < 8; ++kk) {
        const v16bf bfrag = *(const v16bf*)(buf + kk * 1024 + lane * 32);  // ds_load x2
        acc = WMMA_BF16(a[kk], bfrag, acc);
      }

      // acc[r] = P[m0+8g+r, nt*16+n]; weight by e2 (f32 from LDS), fold in.
      const float* wp = e2w + ((size_t)nt * 32 + lane) * 8;
      const float4v w0 = *(const float4v*)wp;
      const float4v w1 = *(const float4v*)(wp + 4);
#pragma unroll
      for (int r = 0; r < 4; ++r) {
        acc_t[r]     += acc[r]     * w0[r];
        acc_t[4 + r] += acc[4 + r] * w1[r];
      }
    }

    // Finish k-slice: reduce over the 16 column-lanes, keep on lane col==k.
#pragma unroll
    for (int r = 0; r < 8; ++r) {
      float t = acc_t[r];
      t += __shfl_xor(t, 1, 32);
      t += __shfl_xor(t, 2, 32);
      t += __shfl_xor(t, 4, 32);
      t += __shfl_xor(t, 8, 32);
      tsel[r] = (n == k) ? t : tsel[r];
    }
  }

  // -------- Epilogue: biases + tanh, store [B, 32] ------------------------
  const float b1n = b1[n];
  const float b2n = b2[n];
#pragma unroll
  for (int r = 0; r < 8; ++r) {
    const int m = m0 + 8 * g + r;
    out[(size_t)m * 32 + n]      = tanhf(accM[r] + b1n);
    out[(size_t)m * 32 + 16 + n] = tanhf(tsel[r] + accF[r] + b2n);
  }
}

// ---------------------------------------------------------------------------
extern "C" void kernel_launch(void* const* d_in, const int* in_sizes, int n_in,
                              void* d_out, int out_size, void* d_ws, size_t ws_size,
                              hipStream_t stream) {
  const float* e1 = (const float*)d_in[0];
  const float* e2 = (const float*)d_in[1];
  const float* W  = (const float*)d_in[2];
  const float* V  = (const float*)d_in[3];
  const float* G  = (const float*)d_in[4];
  const float* b1 = (const float*)d_in[5];
  const float* b2 = (const float*)d_in[6];
  float* out = (float*)d_out;

  __bf16* pw = (__bf16*)d_ws;            // 1,048,576 bf16 = 2 MB
  __bf16* pv = pw + (1u << 20);          // 8192 bf16
  __bf16* pg = pv + 8192;                // 8192 bf16

  hipLaunchKernelGGL(pack_w_kernel,  dim3(4096), dim3(256), 0, stream, W, pw);
  hipLaunchKernelGGL(pack_vg_kernel, dim3(32),   dim3(256), 0, stream, V, G, pv, pg);

  const size_t smem_bytes = 2 * STAGE_BYTES + 8 * 16 * 32 * 8 * sizeof(float); // 144 KB
  hipLaunchKernelGGL(ntn_main_kernel, dim3(65536 / 128), dim3(256), smem_bytes, stream,
                     e1, e2, pw, pv, pg, b1, b2, out);
}
